// CustomNet_30966714204481
// MI455X (gfx1250) — matrix operand order repaired and checked by reading, not compiled, not run on hardware
//
#include <hip/hip_runtime.h>
#include <hip/hip_bf16.h>

typedef __attribute__((ext_vector_type(16))) _Float16 v16h;
typedef __attribute__((ext_vector_type(8)))  float    v8f;

#define BB   16      // batch
#define TT   4096    // timesteps
#define IN   40      // input features
#define HH   100     // hidden
#define G4   400     // 4*H
#define NTIL 25      // 400/16 column tiles
#define NWAV 8       // waves per block

// ws layout (bytes)
#define XG_BYTES   ((size_t)TT * NTIL * 32 * 8 * 4)    // 104,857,600 (fragment order)
#define XA_BYTES   ((size_t)TT * 2 * 32 * 16 * 2)      // 8,388,608
#define WIHB_BYTES ((size_t)NTIL * 2 * 32 * 16 * 2)    // 51,200
#define WHHB_BYTES ((size_t)NTIL * 4 * 32 * 16 * 2)    // 102,400

// -----------------------------------------------------------------------------
// Prep 1: swizzle x (f32 [B][T][IN]) into WMMA A-fragment order (f16),
// K padded 40 -> 64. Layout: xA[((t*2 + ks)*32 + lane)*16 + j]
// -----------------------------------------------------------------------------
__global__ __launch_bounds__(256)
void prep_xA_kernel(const float* __restrict__ x, _Float16* __restrict__ xA)
{
    int idx = blockIdx.x * 256 + threadIdx.x;     // < TT*2*32*16
    int j  = idx & 15;
    int L  = (idx >> 4) & 31;
    int ks = (idx >> 9) & 1;
    int t  = idx >> 10;
    int b  = L & 15;
    int klocal = (L < 16) ? ((j < 8) ? j : j + 8)
                          : ((j < 8) ? j + 8 : j + 16);
    int k  = ks * 32 + klocal;
    int kc = (k < IN) ? k : 0;                    // clamp -> unconditional load
    float v = x[((size_t)b * TT + t) * IN + kc];
    xA[idx] = (k < IN) ? (_Float16)v : (_Float16)0.0f;
}

// -----------------------------------------------------------------------------
// Prep 2: swizzle a [G4][Kdim] f32 weight into WMMA B-fragment order (f16).
// Layout: out[((nt*nks + ks)*32 + lane)*16 + j]; col = lane&15,
// k = ks*32 + (j<8 ? j : j+8) + 8*(lane>=16)
// -----------------------------------------------------------------------------
__global__ __launch_bounds__(256)
void prep_wB_kernel(const float* __restrict__ W, _Float16* __restrict__ out,
                    int Kdim, int nks, int total)
{
    int idx = blockIdx.x * 256 + threadIdx.x;
    if (idx >= total) return;
    int j  = idx & 15;
    int L  = (idx >> 4) & 31;
    int q  = idx >> 9;                // nt*nks + ks
    int ks = q % nks;
    int nt = q / nks;
    int n  = nt * 16 + (L & 15);
    int k  = ks * 32 + ((j < 8) ? j : j + 8) + ((L >> 4) << 3);
    int kc = (k < Kdim) ? k : 0;
    float v = W[n * Kdim + kc];
    out[idx] = (k < Kdim) ? (_Float16)v : (_Float16)0.0f;
}

// -----------------------------------------------------------------------------
// Kernel 1: xgF = x·W_ih^T + (b_ih + b_hh), output in C-fragment order:
//   xgF[(((size_t)t*NTIL + nt)*32 + lane)*8 + v]
// One block per timestep, 8 waves cover the 25 column tiles; accumulator is
// seeded with the bias; epilogue is two lane-contiguous b128 stores per tile.
// -----------------------------------------------------------------------------
__global__ __launch_bounds__(256)
void xg_gemm_kernel(const _Float16* __restrict__ xA,
                    const _Float16* __restrict__ wihB,
                    const float* __restrict__ b_ih,
                    const float* __restrict__ b_hh,
                    float* __restrict__ xgF)
{
    __shared__ float biasLds[G4];
    const int tid  = threadIdx.x;
    const int wave = tid >> 5;
    const int lane = tid & 31;
    const int lm   = lane & 15;
    const int t    = blockIdx.x;

    for (int i = tid; i < G4; i += 256) biasLds[i] = b_ih[i] + b_hh[i];
    __syncthreads();

    v16h a0 = *(const v16h*)(xA + ((size_t)t * 2 + 0) * 512 + lane * 16);
    v16h a1 = *(const v16h*)(xA + ((size_t)t * 2 + 1) * 512 + lane * 16);

    for (int nt = wave; nt < NTIL; nt += NWAV) {
        const float bias = biasLds[nt * 16 + lm];
        v8f c;
        #pragma unroll
        for (int v = 0; v < 8; ++v) c[v] = bias;

        v16h b0 = *(const v16h*)(wihB + (nt * 2 + 0) * 512 + lane * 16);
        v16h b1 = *(const v16h*)(wihB + (nt * 2 + 1) * 512 + lane * 16);
        c = __builtin_amdgcn_wmma_f32_16x16x32_f16(false, a0, false, b0,
                                                   (short)0, c, false, false);
        c = __builtin_amdgcn_wmma_f32_16x16x32_f16(false, a1, false, b1,
                                                   (short)0, c, false, false);

        // store fragment-ordered: 32 contiguous bytes per lane
        *(v8f*)(xgF + (((size_t)t * NTIL + nt) * 32 + lane) * 8) = c;
    }
}

// -----------------------------------------------------------------------------
// Kernel 2: persistent single-workgroup sequential LSTM + head.
// Accumulators seeded straight from fragment-ordered xgF (one v8f load/tile).
// -----------------------------------------------------------------------------
__global__ __launch_bounds__(256)
void lstm_seq_kernel(const float* __restrict__ xgF,
                     const _Float16* __restrict__ whhB_g,
                     const float* __restrict__ W1,
                     const float* __restrict__ b1,
                     const float* __restrict__ gamma,
                     const float* __restrict__ beta,
                     const float* __restrict__ W2,
                     const float* __restrict__ b2,
                     float* __restrict__ out)
{
    __shared__ _Float16 whhB[NTIL * 4 * 32 * 16]; // 100 KB: B fragments
    __shared__ _Float16 hA[4 * 32 * 16];          // 4 KB:  A fragments (h)
    __shared__ float    zbuf[BB * G4];            // 25.6 KB: z / y staging
    __shared__ float    hbuf[BB * HH];            // h (f32)
    __shared__ float    cbuf[BB * HH];            // c (f32)
    __shared__ float    bnscale[HH];
    __shared__ float    bnshift[HH];

    const int tid  = threadIdx.x;
    const int wave = tid >> 5;
    const int lane = tid & 31;
    const int hiHalf = lane >> 4;
    const int lm   = lane & 15;

    // ---- init: bulk-copy prepped W_hh fragments into LDS ----
    {
        const uint4* src = (const uint4*)whhB_g;
        uint4* dst = (uint4*)whhB;
        for (int i = tid; i < (int)(WHHB_BYTES / 16); i += 256) dst[i] = src[i];
    }
    for (int i = tid; i < 4 * 32 * 16; i += 256) hA[i] = (_Float16)0.0f;
    for (int i = tid; i < BB * HH; i += 256) { hbuf[i] = 0.0f; cbuf[i] = 0.0f; }
    __syncthreads();

    // ---- sequential timestep loop ----
    for (int t = 0; t < TT; ++t) {
        v16h a0 = *(const v16h*)(hA + (0 * 32 + lane) * 16);
        v16h a1 = *(const v16h*)(hA + (1 * 32 + lane) * 16);
        v16h a2 = *(const v16h*)(hA + (2 * 32 + lane) * 16);
        v16h a3 = *(const v16h*)(hA + (3 * 32 + lane) * 16);

        for (int nt = wave; nt < NTIL; nt += NWAV) {
            const int col = nt * 16 + lm;
            // seed accumulator: one aligned 32-byte fragment load per lane
            v8f c = *(const v8f*)(xgF + (((size_t)t * NTIL + nt) * 32 + lane) * 8);

            #pragma unroll
            for (int ks = 0; ks < 4; ++ks) {
                v16h bf = *(const v16h*)(whhB + ((nt * 4 + ks) * 32 + lane) * 16);
                v16h af = (ks == 0) ? a0 : (ks == 1) ? a1 : (ks == 2) ? a2 : a3;
                c = __builtin_amdgcn_wmma_f32_16x16x32_f16(false, af, false, bf,
                                                           (short)0, c, false, false);
            }
            #pragma unroll
            for (int v = 0; v < 8; ++v)
                zbuf[(v + (hiHalf << 3)) * G4 + col] = c[v];
        }

        // prefetch next step's xgF chunk while gates run
        {
            int tn = (t + 1 < TT) ? (t + 1) : t;
            __builtin_prefetch(xgF + (size_t)tn * NTIL * 256 + tid * 25, 0, 3);
        }
        __syncthreads();

        // elementwise gates: c = f*c + i*g ; h = o*tanh(c)
        for (int idx = tid; idx < BB * HH; idx += 256) {
            int b_  = idx / HH;
            int hid = idx % HH;
            const float* zr = zbuf + b_ * G4;
            float zi = zr[hid];
            float zf = zr[HH + hid];
            float zg = zr[2 * HH + hid];
            float zo = zr[3 * HH + hid];
            float ig = 1.0f / (1.0f + __expf(-zi));
            float fg = 1.0f / (1.0f + __expf(-zf));
            float gg = tanhf(zg);
            float og = 1.0f / (1.0f + __expf(-zo));
            float cc = fg * cbuf[idx] + ig * gg;
            float hh = og * tanhf(cc);
            cbuf[idx] = cc;
            hbuf[idx] = hh;
            // scatter into WMMA A-fragment layout
            int k  = hid;
            int ks = k >> 5;
            int kk = k & 31;
            int L, j;
            if      (kk < 8)  { L = b_;      j = kk;      }
            else if (kk < 16) { L = b_ + 16; j = kk - 8;  }
            else if (kk < 24) { L = b_;      j = kk - 8;  }
            else              { L = b_ + 16; j = kk - 16; }
            hA[(ks * 32 + L) * 16 + j] = (_Float16)hh;
        }
        __syncthreads();
    }

    // ---- head: y = h@W1^T + b1 ----
    for (int idx = tid; idx < BB * HH; idx += 256) {
        int b_ = idx / HH;
        int j  = idx % HH;
        float s = b1[j];
        const float* hr = hbuf + b_ * HH;
        const float* wr = W1 + j * HH;
        for (int k = 0; k < HH; ++k) s += hr[k] * wr[k];
        zbuf[b_ * HH + j] = s;
    }
    __syncthreads();

    // ---- batchnorm (training mode, biased var over B=16) ----
    if (tid < HH) {
        float mu = 0.0f;
        for (int b_ = 0; b_ < BB; ++b_) mu += zbuf[b_ * HH + tid];
        mu *= (1.0f / BB);
        float var = 0.0f;
        for (int b_ = 0; b_ < BB; ++b_) {
            float d = zbuf[b_ * HH + tid] - mu;
            var += d * d;
        }
        var *= (1.0f / BB);
        float inv = rsqrtf(var + 1e-5f);
        float sc  = gamma[tid] * inv;
        bnscale[tid] = sc;
        bnshift[tid] = beta[tid] - mu * sc;
    }
    __syncthreads();

    // ---- out = BN(y)@W2^T + b2 -> [16, 40] ----
    for (int idx = tid; idx < BB * 40; idx += 256) {
        int b_ = idx / 40;
        int o  = idx % 40;
        float s = b2[o];
        const float* yr = zbuf + b_ * HH;
        const float* wr = W2 + o * HH;
        for (int j = 0; j < HH; ++j)
            s += (yr[j] * bnscale[j] + bnshift[j]) * wr[j];
        out[idx] = s;
    }
}

extern "C" void kernel_launch(void* const* d_in, const int* in_sizes, int n_in,
                              void* d_out, int out_size, void* d_ws, size_t ws_size,
                              hipStream_t stream) {
    const float* x     = (const float*)d_in[0];
    const float* W_ih  = (const float*)d_in[1];
    const float* W_hh  = (const float*)d_in[2];
    const float* b_ih  = (const float*)d_in[3];
    const float* b_hh  = (const float*)d_in[4];
    const float* W1    = (const float*)d_in[5];
    const float* b1    = (const float*)d_in[6];
    const float* gamma = (const float*)d_in[7];
    const float* beta  = (const float*)d_in[8];
    const float* W2    = (const float*)d_in[9];
    const float* b2    = (const float*)d_in[10];
    float* out = (float*)d_out;

    char* ws = (char*)d_ws;
    float*     xgF  = (float*)ws;
    _Float16*  xA   = (_Float16*)(ws + XG_BYTES);
    _Float16*  wihB = (_Float16*)(ws + XG_BYTES + XA_BYTES);
    _Float16*  whhB = (_Float16*)(ws + XG_BYTES + XA_BYTES + WIHB_BYTES);

    // prep: fragment-order f16 buffers
    prep_xA_kernel<<<(TT * 1024) / 256, 256, 0, stream>>>(x, xA);
    prep_wB_kernel<<<(NTIL * 2 * 512 + 255) / 256, 256, 0, stream>>>(
        W_ih, wihB, IN, 2, NTIL * 2 * 512);
    prep_wB_kernel<<<(NTIL * 4 * 512 + 255) / 256, 256, 0, stream>>>(
        W_hh, whhB, HH, 4, NTIL * 4 * 512);

    // main GEMM: xgF = x·W_ih^T + bias (fragment-ordered output)
    xg_gemm_kernel<<<TT, 256, 0, stream>>>(xA, wihB, b_ih, b_hh, xgF);

    // sequential LSTM + head
    lstm_seq_kernel<<<1, 256, 0, stream>>>(xgF, whhB, W1, b1, gamma, beta, W2, b2, out);
}
